// MultiHeadAttention_25881472926482
// MI455X (gfx1250) — matrix operand order repaired and checked by reading, not compiled, not run on hardware
//
#include <hip/hip_runtime.h>
#include <cstdint>
#include <cstddef>

// ---------------- problem constants ----------------
#define D_MODEL   1024
#define NUM_HEADS 16
#define DK        64
#define BATCH     4
#define SEQ       2048
#define M_TOT     (BATCH * SEQ)               // 8192 rows for the big GEMMs
#define NE        ((size_t)M_TOT * D_MODEL)   // 8,388,608 activation elems
#define WE        ((size_t)D_MODEL * D_MODEL) // 1,048,576 weight elems

typedef __bf16 bf16_t;
typedef __bf16 bf16x16 __attribute__((ext_vector_type(16)));
typedef __bf16 bf16x8  __attribute__((ext_vector_type(8)));
typedef __bf16 bf16x4  __attribute__((ext_vector_type(4)));
typedef float  f32x8   __attribute__((ext_vector_type(8)));

// round-to-nearest-even f32 -> bf16 (pure integer ops)
static __device__ __forceinline__ bf16_t f2bf(float f) {
  unsigned u = __builtin_bit_cast(unsigned, f);
  unsigned r = u + 0x7FFFu + ((u >> 16) & 1u);
  unsigned short s = (unsigned short)(r >> 16);
  return __builtin_bit_cast(bf16_t, s);
}

// 16-bit A-operand per ISA layout: lane holds two contiguous 8-elem K-runs.
static __device__ __forceinline__ bf16x16 load_a16(const bf16_t* p0, const bf16_t* p1) {
  bf16x8 lo = *(const bf16x8*)p0;
  bf16x8 hi = *(const bf16x8*)p1;
  return __builtin_shufflevector(lo, hi, 0,1,2,3,4,5,6,7,8,9,10,11,12,13,14,15);
}

static __device__ __forceinline__ f32x8 wmma_bf16(bf16x16 a, bf16x16 b, f32x8 c) {
  return __builtin_amdgcn_wmma_f32_16x16x32_bf16(false, a, false, b, (short)0, c, false, false);
}

// ---------------- elementwise fp32 -> bf16 (x4 vectorized) ----------------
__global__ void cvt_bf16_kernel(const float* __restrict__ in, bf16_t* __restrict__ out, int n4) {
  int i = blockIdx.x * blockDim.x + threadIdx.x;
  if (i < n4) {
    float4 f = ((const float4*)in)[i];
    bf16x4 o;
    o[0] = f2bf(f.x); o[1] = f2bf(f.y); o[2] = f2bf(f.z); o[3] = f2bf(f.w);
    *(bf16x4*)(out + (size_t)i * 4) = o;
  }
}

// W[in][out] fp32 -> WT[out][in] bf16 (GEMM B-operand K becomes contiguous)
__global__ void wtrans_bf16_kernel(const float* __restrict__ W, bf16_t* __restrict__ WT) {
  int i = blockIdx.x * blockDim.x + threadIdx.x;  // over D_MODEL*D_MODEL
  if (i < (int)WE) {
    int r = i / D_MODEL;          // input dim
    int c = i - r * D_MODEL;      // output dim
    WT[(size_t)c * D_MODEL + r] = f2bf(W[i]);
  }
}

// ------- register-blocked bf16 WMMA GEMM: C = A[M,K] * WT[N,K]^T + bias -------
// Each wave computes a 64x32 output block: 4 m-tiles x 2 n-tiles, 8 WMMAs per
// K-step of 32 against 12 b128 loads (24 B per lane per WMMA).
// mode 0: bf16 out, row-major [M][N]
// mode 1: bf16 out, V-transposed layout [b][h][dk][key]
// mode 2: fp32 out (non-temporal), row-major [M][N]
__global__ __launch_bounds__(256) void gemm_bf16_kernel(
    const bf16_t* __restrict__ A, const bf16_t* __restrict__ WT,
    const float* __restrict__ bias, void* __restrict__ out,
    int M, int K, int N, int mode) {
  const int lane = threadIdx.x & 31;
  const int wave = threadIdx.x >> 5;          // 0..7
  const int wm   = wave >> 2;                 // 0..1  (m-direction in block)
  const int wn   = wave & 3;                  // 0..3  (n-direction in block)
  const int lr   = lane & 15;                 // tile row (A) / tile col (B)
  const int hk   = lane >> 4;                 // K half-select

  const int mt0 = blockIdx.x * 8 + wm * 4;    // first of 4 m-tiles (16 rows ea.)
  const int nt0 = blockIdx.y * 8 + wn * 2;    // first of 2 n-tiles (16 cols ea.)

  const bf16_t* arow[4];
#pragma unroll
  for (int mi = 0; mi < 4; ++mi)
    arow[mi] = A + ((size_t)(mt0 + mi) * 16 + lr) * (size_t)K;
  const bf16_t* brow[2];
#pragma unroll
  for (int ni = 0; ni < 2; ++ni)
    brow[ni] = WT + ((size_t)(nt0 + ni) * 16 + lr) * (size_t)K;

  f32x8 acc[4][2] = {};
  for (int k = 0; k < K; k += 32) {
    bf16x16 a[4], b[2];
#pragma unroll
    for (int mi = 0; mi < 4; ++mi)
      a[mi] = load_a16(arow[mi] + k + hk * 8, arow[mi] + k + 16 + hk * 8);
#pragma unroll
    for (int ni = 0; ni < 2; ++ni)
      b[ni] = load_a16(brow[ni] + k + hk * 8, brow[ni] + k + 16 + hk * 8);
    if (k + 256 < K) {                        // stream-ahead (global_prefetch_b8)
      __builtin_prefetch(arow[0] + k + 256, 0, 1);
      __builtin_prefetch(brow[0] + k + 256, 0, 1);
    }
#pragma unroll
    for (int mi = 0; mi < 4; ++mi)
#pragma unroll
      for (int ni = 0; ni < 2; ++ni)
        acc[mi][ni] = wmma_bf16(a[mi], b[ni], acc[mi][ni]);
  }

#pragma unroll
  for (int ni = 0; ni < 2; ++ni) {
    const int nt = nt0 + ni;
    const float bv = bias[nt * 16 + lr];
#pragma unroll
    for (int mi = 0; mi < 4; ++mi) {
      const int mt = mt0 + mi;
#pragma unroll
      for (int r = 0; r < 8; ++r) {
        float val = acc[mi][ni][r] + bv;
        int m = mt * 16 + r + 8 * hk;         // output row
        int n = nt * 16 + lr;                 // output col
        if (mode == 0) {
          ((bf16_t*)out)[(size_t)m * N + n] = f2bf(val);
        } else if (mode == 1) {
          int b_ = m >> 11, key = m & (SEQ - 1);
          int h_ = n >> 6,  dk  = n & (DK - 1);
          ((bf16_t*)out)[(((size_t)(b_ * NUM_HEADS + h_) * DK + dk) * SEQ) + key] = f2bf(val);
        } else {
          __builtin_nontemporal_store(val, (float*)out + (size_t)m * N + n);
        }
      }
    }
  }
}

// ---------------- fused scores + softmax + P*V per (b, h, 16-query tile) ------
// 4 waves / block. Dynamic LDS: 16*2048 fp32 scores | 16*2048 bf16 probs | 16*8 red
__global__ __launch_bounds__(128) void attn_kernel(
    const bf16_t* __restrict__ Qp,   // [b*SEQ + l][h*DK + d] bf16
    const bf16_t* __restrict__ Kp,   // same layout
    const bf16_t* __restrict__ VT,   // [b][h][dk][key] bf16
    bf16_t* __restrict__ ctx,        // [b*SEQ + l][h*DK + d] bf16
    float* __restrict__ wout) {      // [b][h][q][k] fp32  (second output)
  extern __shared__ char smem_raw[];
  float*  s_sc = (float*)smem_raw;                                  // 16 x 2048 f32
  bf16_t* s_pb = (bf16_t*)(smem_raw + 16 * SEQ * sizeof(float));    // 16 x 2048 bf16
  float*  s_rd = (float*)(smem_raw + 16 * SEQ * (sizeof(float) + sizeof(bf16_t)));

  const int qt = blockIdx.x;              // 0..127
  const int bh = blockIdx.y;              // 0..63
  const int b_ = bh >> 4, h_ = bh & (NUM_HEADS - 1);
  const int lane = threadIdx.x & 31;
  const int wave = threadIdx.x >> 5;      // 0..3
  const int lr = lane & 15;
  const int hk = lane >> 4;

  // ---- phase 1: scores = (Q K^T) / sqrt(dk) into LDS (Q tile reused 32x) ----
  const bf16_t* qrow =
      Qp + ((size_t)b_ * SEQ + qt * 16 + lr) * D_MODEL + h_ * DK;
  bf16x16 a0 = load_a16(qrow +      hk * 8, qrow + 16 + hk * 8);   // dk 0..31
  bf16x16 a1 = load_a16(qrow + 32 + hk * 8, qrow + 48 + hk * 8);   // dk 32..63

  for (int t = 0; t < 32; ++t) {
    int ct = wave * 32 + t;               // 16-key column tile
    const bf16_t* krow =
        Kp + ((size_t)b_ * SEQ + ct * 16 + lr) * D_MODEL + h_ * DK;
    bf16x16 b0 = load_a16(krow +      hk * 8, krow + 16 + hk * 8);
    bf16x16 b1 = load_a16(krow + 32 + hk * 8, krow + 48 + hk * 8);
    f32x8 c = {};
    c = wmma_bf16(a0, b0, c);
    c = wmma_bf16(a1, b1, c);
#pragma unroll
    for (int r = 0; r < 8; ++r)
      s_sc[(r + 8 * hk) * SEQ + ct * 16 + lr] = c[r] * 0.125f;  // 1/sqrt(64)
  }
  __syncthreads();

  // ---- phase 2: row softmax; fp32 weights to d_out (NT), bf16 probs to LDS ----
  const int row = threadIdx.x >> 3;       // 16 rows, 8 threads each
  const int sub = threadIdx.x & 7;
  float* srow = s_sc + row * SEQ;

  float mx = -3.4028235e38f;
  for (int c = sub; c < SEQ; c += 8) mx = fmaxf(mx, srow[c]);
  s_rd[row * 8 + sub] = mx;
  __syncthreads();
  if (sub == 0) {
    float mm = s_rd[row * 8];
#pragma unroll
    for (int j = 1; j < 8; ++j) mm = fmaxf(mm, s_rd[row * 8 + j]);
    s_rd[row * 8] = mm;
  }
  __syncthreads();
  const float mm = s_rd[row * 8];

  float sum = 0.0f;
  for (int c = sub; c < SEQ; c += 8) {
    float e = __expf(srow[c] - mm);
    srow[c] = e;
    sum += e;
  }
  __syncthreads();
  s_rd[row * 8 + sub] = sum;
  __syncthreads();
  if (sub == 0) {
    float ss = 0.0f;
#pragma unroll
    for (int j = 0; j < 8; ++j) ss += s_rd[row * 8 + j];
    s_rd[row * 8] = ss;
  }
  __syncthreads();
  const float inv = 1.0f / s_rd[row * 8];

  float* wrow = wout + (((size_t)bh * SEQ) + qt * 16 + row) * SEQ;
  for (int c = sub; c < SEQ; c += 8) {
    float p = srow[c] * inv;
    __builtin_nontemporal_store(p, wrow + c);  // 1.07 GB write-once stream
    s_pb[row * SEQ + c] = f2bf(p);             // bf16 probs for P*V
  }
  __syncthreads();

  // ---- phase 3: context = P(16x2048) * V(2048x64); wave owns one dk 16-chunk ----
  f32x8 acc = {};
  const bf16_t* prow = s_pb + lr * SEQ;
  const bf16_t* vrow =
      VT + ((size_t)(b_ * NUM_HEADS + h_) * DK + wave * 16 + lr) * SEQ;
  for (int kk = 0; kk < SEQ; kk += 32) {
    bf16x16 a = load_a16(prow + kk + hk * 8, prow + kk + 16 + hk * 8);
    bf16x16 v = load_a16(vrow + kk + hk * 8, vrow + kk + 16 + hk * 8);
    acc = wmma_bf16(a, v, acc);
  }
#pragma unroll
  for (int r = 0; r < 8; ++r) {
    int q = qt * 16 + r + 8 * hk;
    int n = h_ * DK + wave * 16 + lr;
    ctx[((size_t)b_ * SEQ + q) * D_MODEL + n] = f2bf(acc[r]);
  }
}

// ---------------- host-side orchestration ----------------
extern "C" void kernel_launch(void* const* d_in, const int* in_sizes, int n_in,
                              void* d_out, int out_size, void* d_ws, size_t ws_size,
                              hipStream_t stream) {
  const float* q   = (const float*)d_in[0];
  const float* k   = (const float*)d_in[1];
  const float* v   = (const float*)d_in[2];
  const float* W_q = (const float*)d_in[3];
  const float* b_q = (const float*)d_in[4];
  const float* W_k = (const float*)d_in[5];
  const float* b_k = (const float*)d_in[6];
  const float* W_v = (const float*)d_in[7];
  const float* b_v = (const float*)d_in[8];
  const float* W_o = (const float*)d_in[9];
  const float* b_o = (const float*)d_in[10];

  float* out_f   = (float*)d_out;                     // [4,2048,1024]
  float* weights = out_f + NE;                        // [4,16,2048,2048]

  // workspace carve-up (bf16 elements)
  bf16_t* w   = (bf16_t*)d_ws;
  bf16_t* qb  = w;            // bf16 activations
  bf16_t* kb  = qb  + NE;
  bf16_t* vb  = kb  + NE;
  bf16_t* WqT = vb  + NE;     // bf16 transposed weights [out][in]
  bf16_t* WkT = WqT + WE;
  bf16_t* WvT = WkT + WE;
  bf16_t* WoT = WvT + WE;
  bf16_t* Qp  = WoT + WE;     // projected Q/K (row-major), V (transposed), ctx
  bf16_t* Kp  = Qp  + NE;
  bf16_t* VT  = Kp  + NE;
  bf16_t* ctx = VT  + NE;

  // 1) fp32 -> bf16 activations (x4 vectorized)
  {
    int n4 = (int)(NE / 4);
    dim3 blk(256), grd((unsigned)(n4 / 256));
    cvt_bf16_kernel<<<grd, blk, 0, stream>>>(q, qb, n4);
    cvt_bf16_kernel<<<grd, blk, 0, stream>>>(k, kb, n4);
    cvt_bf16_kernel<<<grd, blk, 0, stream>>>(v, vb, n4);
  }
  // 2) weight transpose + convert
  {
    dim3 blk(256), grd((unsigned)(WE / 256));
    wtrans_bf16_kernel<<<grd, blk, 0, stream>>>(W_q, WqT);
    wtrans_bf16_kernel<<<grd, blk, 0, stream>>>(W_k, WkT);
    wtrans_bf16_kernel<<<grd, blk, 0, stream>>>(W_v, WvT);
    wtrans_bf16_kernel<<<grd, blk, 0, stream>>>(W_o, WoT);
  }
  // 3) Q/K/V projections (blocked WMMA GEMMs); V lands pre-transposed for P*V
  {
    dim3 blk(256), grd(M_TOT / 128, D_MODEL / 128);   // (64, 8)
    gemm_bf16_kernel<<<grd, blk, 0, stream>>>(qb, WqT, b_q, Qp, M_TOT, D_MODEL, D_MODEL, 0);
    gemm_bf16_kernel<<<grd, blk, 0, stream>>>(kb, WkT, b_k, Kp, M_TOT, D_MODEL, D_MODEL, 0);
    gemm_bf16_kernel<<<grd, blk, 0, stream>>>(vb, WvT, b_v, VT, M_TOT, D_MODEL, D_MODEL, 1);
  }
  // 4) fused attention: scores -> softmax (weights out) -> P*V
  {
    dim3 blk(128), grd(SEQ / 16, BATCH * NUM_HEADS);
    size_t shmem = 16 * SEQ * sizeof(float)     // fp32 scores
                 + 16 * SEQ * sizeof(bf16_t)    // bf16 probs
                 + 16 * 8  * sizeof(float);     // reductions
    attn_kernel<<<grd, blk, shmem, stream>>>(Qp, Kp, VT, ctx, weights);
  }
  // 5) output projection (fp32 result into d_out)
  {
    dim3 blk(256), grd(M_TOT / 128, D_MODEL / 128);
    gemm_bf16_kernel<<<grd, blk, 0, stream>>>(ctx, WoT, b_o, out_f, M_TOT, D_MODEL, D_MODEL, 2);
  }
}
